// GCNLPAConv_1005022347778
// MI455X (gfx1250) — matrix operand order repaired and checked by reading, not compiled, not run on hardware
//
#include <hip/hip_runtime.h>
#include <hip/hip_bf16.h>

typedef __attribute__((ext_vector_type(16))) _Float16 v16h;
typedef __attribute__((ext_vector_type(8)))  float    v8f;

#define N_ROWS 8192
#define D_IN   128
#define D_OUT  128
#define N_CLS  64
#define B_COLS (D_OUT + N_CLS)   // 192 fused output columns

// ---------------------------------------------------------------------------
// Kernel 1: support = x @ weight  (fp32 VALU, tiny: 0.27 GFLOP), stored as
// f16 COLUMN-MAJOR into Bcol[j][r] so kernel 3's B-fragment loads are
// contiguous 32B per lane.
// ---------------------------------------------------------------------------
__global__ __launch_bounds__(128)
void build_support_k(const float* __restrict__ x, const float* __restrict__ w,
                     _Float16* __restrict__ Bcol) {
  __shared__ float xsT[D_IN][33];          // transposed x tile, padded vs bank conflicts
  const int r0 = blockIdx.x * 32;
  const int t  = threadIdx.x;              // t = k index on load, j (output col) on compute
  #pragma unroll 4
  for (int i = 0; i < 32; ++i)
    xsT[t][i] = x[(size_t)(r0 + i) * D_IN + t];
  __syncthreads();

  float acc[32];
  #pragma unroll
  for (int r = 0; r < 32; ++r) acc[r] = 0.0f;
  for (int k = 0; k < D_IN; ++k) {
    const float wk = w[k * D_OUT + t];     // coalesced across threads
    #pragma unroll
    for (int r = 0; r < 32; ++r) acc[r] += xsT[k][r] * wk;
  }
  _Float16* col = Bcol + (size_t)t * N_ROWS + r0;
  #pragma unroll
  for (int r = 0; r < 32; ++r) col[r] = (_Float16)acc[r];
}

// ---------------------------------------------------------------------------
// Kernel 2: pack y (f32 [8192,64]) into Bcol columns 128..191 as f16.
// ---------------------------------------------------------------------------
__global__ __launch_bounds__(256)
void pack_y_k(const float* __restrict__ y, _Float16* __restrict__ Bcol) {
  const int idx = blockIdx.x * 256 + threadIdx.x;   // [0, 8192*64)
  const int r = idx & (N_ROWS - 1);
  const int c = idx >> 13;
  Bcol[(size_t)(D_OUT + c) * N_ROWS + r] = (_Float16)y[(size_t)r * N_CLS + c];
}

// ---------------------------------------------------------------------------
// Kernel 3: fused masked softmax (unnormalized) + [A@support | A@y] + bias.
// One block = 16 output rows. 4 waves split K=8192 (2048 each); partial
// accumulators + row sums merged deterministically through LDS.
// ---------------------------------------------------------------------------
#define PEXP(av, mv) (((av) > 0) ? __expf(mv) : 0.0f)

__global__ __launch_bounds__(128)
void fused_softmax_spmm_k(const int* __restrict__ adj,
                          const float* __restrict__ msk,
                          const _Float16* __restrict__ Bcol,
                          const float* __restrict__ bias,
                          float* __restrict__ out1,      // [8192,128]
                          float* __restrict__ out2) {    // [8192, 64]
  __shared__ float redAcc[4][16 * B_COLS];   // per-wave partial tiles
  __shared__ float redL[4][16];              // per-wave partial row sums

  const int tid  = threadIdx.x;
  const int lane = tid & 31;
  const int wave = tid >> 5;
  const int M    = lane & 15;                // row of A-frag / col N of B-frag
  const int h    = lane >> 4;                // lane half selects K sub-runs
  const int m0   = blockIdx.x * 16;
  const size_t rowbase = (size_t)(m0 + M) * N_ROWS;
  const int*   adjr = adj + rowbase;
  const float* mskr = msk + rowbase;

  v8f acc[12];
  const v8f vzero = {};
  #pragma unroll
  for (int t = 0; t < 12; ++t) acc[t] = vzero;
  float lsum = 0.0f;

  const int kbeg = wave * (N_ROWS / 4);
  const int kend = kbeg + (N_ROWS / 4);
  for (int k = kbeg; k < kend; k += 32) {
    // A-fragment K runs (contiguous, matching B):
    // halfs[0..7]  = K = 16h + 0..7 ; halfs[8..15] = K = 16h + 8 + 0..7
    const int kA0 = k + 16 * h;
    const int kA1 = k + 16 * h + 8;
    const int4   ia = *(const int4*)  (adjr + kA0);
    const int4   ib = *(const int4*)  (adjr + kA0 + 4);
    const int4   ic = *(const int4*)  (adjr + kA1);
    const int4   id = *(const int4*)  (adjr + kA1 + 4);
    const float4 fa = *(const float4*)(mskr + kA0);
    const float4 fb = *(const float4*)(mskr + kA0 + 4);
    const float4 fc = *(const float4*)(mskr + kA1);
    const float4 fd = *(const float4*)(mskr + kA1 + 4);
    if (k + 32 < kend) {                    // pull next chunk toward caches
      __builtin_prefetch(adjr + k + 32, 0, 1);
      __builtin_prefetch(mskr + k + 32, 0, 1);
    }

    const float p0  = PEXP(ia.x, fa.x), p1  = PEXP(ia.y, fa.y);
    const float p2  = PEXP(ia.z, fa.z), p3  = PEXP(ia.w, fa.w);
    const float p4  = PEXP(ib.x, fb.x), p5  = PEXP(ib.y, fb.y);
    const float p6  = PEXP(ib.z, fb.z), p7  = PEXP(ib.w, fb.w);
    const float p8  = PEXP(ic.x, fc.x), p9  = PEXP(ic.y, fc.y);
    const float p10 = PEXP(ic.z, fc.z), p11 = PEXP(ic.w, fc.w);
    const float p12 = PEXP(id.x, fd.x), p13 = PEXP(id.y, fd.y);
    const float p14 = PEXP(id.z, fd.z), p15 = PEXP(id.w, fd.w);

    lsum += (((p0 + p1) + (p2 + p3)) + ((p4 + p5) + (p6 + p7)))
          + (((p8 + p9) + (p10 + p11)) + ((p12 + p13) + (p14 + p15)));

    v16h A;
    A[0]  = (_Float16)p0;  A[1]  = (_Float16)p1;  A[2]  = (_Float16)p2;  A[3]  = (_Float16)p3;
    A[4]  = (_Float16)p4;  A[5]  = (_Float16)p5;  A[6]  = (_Float16)p6;  A[7]  = (_Float16)p7;
    A[8]  = (_Float16)p8;  A[9]  = (_Float16)p9;  A[10] = (_Float16)p10; A[11] = (_Float16)p11;
    A[12] = (_Float16)p12; A[13] = (_Float16)p13; A[14] = (_Float16)p14; A[15] = (_Float16)p15;

    // B-fragment: lane holds column N=M, halfs[i] = B[k + 16h + i][N]
    // -> contiguous 32B in column-major Bcol, 32B aligned.
    const int kb = k + 16 * h;
    #pragma unroll
    for (int t = 0; t < 12; ++t) {
      const v16h Bf = *(const v16h*)(Bcol + (((size_t)(t * 16 + M)) << 13) + kb);
      acc[t] = __builtin_amdgcn_wmma_f32_16x16x32_f16(
          /*neg_a=*/false, A, /*neg_b=*/false, Bf,
          /*c_mod=*/(short)0, acc[t], /*reuse_a=*/false, /*reuse_b=*/false);
    }
  }

  // Each (row, col) element lives in exactly one lane per wave -> plain stores
  // into this wave's LDS slab (deterministic; no float atomics).
  #pragma unroll
  for (int t = 0; t < 12; ++t) {
    const int col = t * 16 + M;
    #pragma unroll
    for (int r = 0; r < 8; ++r)
      redAcc[wave][(r + 8 * h) * B_COLS + col] = acc[t][r];
  }
  // Row sum: lanes M and M+16 hold disjoint K halves; merge then one lane writes.
  float lrow = lsum + __shfl_xor(lsum, 16, 32);
  if (h == 0) redL[wave][M] = lrow;
  __syncthreads();

  // Deterministic merge of the 4 K-partials, divide by row sum, add bias.
  for (int i = tid; i < 16 * B_COLS; i += 128) {
    const int row = i / B_COLS;
    const int col = i - row * B_COLS;
    const float s = (redAcc[0][i] + redAcc[1][i]) + (redAcc[2][i] + redAcc[3][i]);
    const float l = (redL[0][row] + redL[1][row]) + (redL[2][row] + redL[3][row]);
    const float v = s / l;
    const int gr = m0 + row;
    if (col < D_OUT) out1[(size_t)gr * D_OUT + col] = v + bias[col];
    else             out2[(size_t)gr * N_CLS + (col - D_OUT)] = v;
  }
}

// ---------------------------------------------------------------------------
extern "C" void kernel_launch(void* const* d_in, const int* in_sizes, int n_in,
                              void* d_out, int out_size, void* d_ws, size_t ws_size,
                              hipStream_t stream) {
  const float* x    = (const float*)d_in[0];
  const float* y    = (const float*)d_in[1];
  const float* msk  = (const float*)d_in[2];
  const int*   adj  = (const int*)  d_in[3];
  const float* w    = (const float*)d_in[4];
  const float* bias = (const float*)d_in[5];
  float* out1 = (float*)d_out;
  float* out2 = out1 + (size_t)N_ROWS * D_OUT;

  _Float16* Bcol = (_Float16*)d_ws;   // [192][8192] f16 column-major, 3 MB

  build_support_k<<<N_ROWS / 32, 128, 0, stream>>>(x, w, Bcol);
  pack_y_k<<<(N_ROWS * N_CLS) / 256, 256, 0, stream>>>(y, Bcol);
  fused_softmax_spmm_k<<<N_ROWS / 16, 128, 0, stream>>>(adj, msk, Bcol, bias,
                                                        out1, out2);
}
